// NeuralCDE_21809843929416
// MI455X (gfx1250) — matrix operand order repaired and checked by reading, not compiled, not run on hardware
//
#include <hip/hip_runtime.h>
#include <hip/hip_bf16.h>
#include <math.h>

// Neural CDE forward on gfx1250 (MI455X), bf16 WMMA + fp32 state.
// B=256, T=128, C=32, H=512, O=8, NSUB=2 (ds=0.5).
//
// - Each wave owns ONE 16x16 output tile (8 waves/block, 64 blocks => 512
//   waves per GEMM stage); row-half wave pairs share B-fragments via WGP$.
// - Dual accumulator chains (even/odd k-tiles) so each wave exposes two
//   independent WMMA dependence chains to the matrix pipe.

typedef __attribute__((ext_vector_type(16))) __bf16 v16bf;
typedef __attribute__((ext_vector_type(8)))  float  v8f;

#define B_SZ 256
#define T_SZ 128
#define TM1  127
#define C_SZ 32
#define H_SZ 512
#define O_SZ 8

__device__ __forceinline__ unsigned short f2bf(float f) {
  unsigned u = __float_as_uint(f);
  unsigned r = ((u >> 16) & 1u) + 0x7FFFu;   // round-to-nearest-even
  return (unsigned short)((u + r) >> 16);
}

union FragB { uint4 q[2]; v16bf v; };
union Acc8  { v8f v; float f[8]; };

#define WMMA_BF16(A, Bm, Cm) \
  __builtin_amdgcn_wmma_f32_16x16x32_bf16(false, (A), false, (Bm), (short)0, (Cm), false, false)

// ---------------------------------------------------------------------------
// Weight packing: B-fragments for v_wmma_f32_16x16x32_bf16 in this code's contiguous k order (A uses the same).
// B is 32(K)x16(N): lanes 0-15 hold K=0..15, lanes 16-31 hold K=16..31,
// VGPR v holds K={2v, 2v+1} packed. Stored lane-contiguous (8 dwords/lane).
// ---------------------------------------------------------------------------
__global__ void pack_w1_kernel(const float* __restrict__ W1, unsigned* __restrict__ W1p) {
  int id = blockIdx.x * blockDim.x + threadIdx.x;       // H*H/2 = 131072
  if (id >= (H_SZ * H_SZ / 2)) return;
  int v = id & 7, lane = (id >> 3) & 31, kt = (id >> 8) & 15, jt = id >> 12;
  int half = lane >> 4, n = lane & 15;
  int j  = jt * 16 + n;
  int k0 = kt * 32 + half * 16 + 2 * v;                 // B1[k,j] = W1[j,k]
  unsigned short lo = f2bf(W1[j * H_SZ + k0]);
  unsigned short hi = f2bf(W1[j * H_SZ + k0 + 1]);
  W1p[id] = (unsigned)lo | ((unsigned)hi << 16);
}

__global__ void pack_w2_kernel(const float* __restrict__ W2, const float* __restrict__ Wv,
                               unsigned* __restrict__ W2p) {
  int id = blockIdx.x * blockDim.x + threadIdx.x;       // C*H*H/2 = 4194304
  if (id >= (C_SZ * H_SZ * H_SZ / 2)) return;
  int v = id & 7, lane = (id >> 3) & 31, kt = (id >> 8) & 15;
  int hht = (id >> 12) & 31, c = id >> 17;
  int half = lane >> 4, n = lane & 15;
  int hh = hht * 16 + n;
  int k0 = kt * 32 + half * 16 + 2 * v;                 // B2c[k,hh] = Wv[c]*W2[hh*C+c, k]
  float wv = Wv[c];
  unsigned short lo = f2bf(wv * W2[(hh * C_SZ + c) * H_SZ + k0]);
  unsigned short hi = f2bf(wv * W2[(hh * C_SZ + c) * H_SZ + k0 + 1]);
  W2p[id] = (unsigned)lo | ((unsigned)hi << 16);
}

// z0 = ca[:,0] @ Wi^T + bi  (tiny, fp32)
__global__ void init_z0_kernel(const float* __restrict__ ca, const float* __restrict__ Wi,
                               const float* __restrict__ bi,
                               float* __restrict__ z, unsigned short* __restrict__ zin) {
  int id = blockIdx.x * blockDim.x + threadIdx.x;       // B*H = 131072
  if (id >= B_SZ * H_SZ) return;
  int b = id >> 9, j = id & 511;
  float acc = bi[j];
  const float* car = ca + (size_t)b * TM1 * C_SZ;       // t = 0
  #pragma unroll
  for (int c = 0; c < C_SZ; ++c) acc += car[c] * Wi[j * C_SZ + c];
  z[id]   = acc;
  zin[id] = f2bf(acc);
}

// ---------------------------------------------------------------------------
// GEMM1: h = ELU(zin @ W1^T + bias1), bf16 out.
// grid (8,8): block = 32 b-rows x 64 j-cols, 8 waves, 1 tile per wave.
// ---------------------------------------------------------------------------
__global__ __launch_bounds__(256) void gemm1_kernel(
    const unsigned short* __restrict__ zin, const uint4* __restrict__ W1p,
    const float* __restrict__ bias1, unsigned short* __restrict__ hbf) {
  __shared__ __align__(16) unsigned short zt[32 * 520];   // padded stride 1040B (16B aligned)
  int bg = blockIdx.x, jg = blockIdx.y, tid = threadIdx.x;

  const uint4* src = (const uint4*)zin + (size_t)bg * 32 * 64;
  uint4* dst = (uint4*)zt;
  for (int i = tid; i < 2048; i += 256) {
    int r = i >> 6, cu = i & 63;
    dst[r * 65 + cu] = src[r * 64 + cu];
  }
  __syncthreads();

  int lane = tid & 31, wave = tid >> 5;
  int wj = wave & 3, wb = wave >> 2;
  int half = lane >> 4, n = lane & 15;
  int jt = jg * 4 + wj;
  const uint4* lq = (const uint4*)zt;

  v8f zf = {0.f,0.f,0.f,0.f,0.f,0.f,0.f,0.f};
  Acc8 aE, aO; aE.v = zf; aO.v = zf;                     // dual chains
  int arow = wb * 16 * 65;                               // row-half offset (uint4)
  #pragma unroll
  for (int kt = 0; kt < 16; kt += 2) {
    FragB bfr0, bfr1;
    const uint4* wbp0 = W1p + (((jt * 16 + kt)     * 32 + lane) << 1);
    const uint4* wbp1 = W1p + (((jt * 16 + kt + 1) * 32 + lane) << 1);
    bfr0.q[0] = wbp0[0]; bfr0.q[1] = wbp0[1];
    bfr1.q[0] = wbp1[0]; bfr1.q[1] = wbp1[1];
    FragB af0, af1;
    int ab = arow + n * 65 + kt * 4 + half * 2;          // A-frag: 2x ds_load_b128
    af0.q[0] = lq[ab];     af0.q[1] = lq[ab + 1];
    af1.q[0] = lq[ab + 4]; af1.q[1] = lq[ab + 5];
    aE.v = WMMA_BF16(af0.v, bfr0.v, aE.v);
    aO.v = WMMA_BF16(af1.v, bfr1.v, aO.v);
  }

  int j = jt * 16 + n;
  float bias = bias1[j];
  #pragma unroll
  for (int v = 0; v < 8; ++v) {
    int m = v + half * 8;                                // D-frag row
    float x0 = aE.f[v] + aO.f[v] + bias;
    x0 = x0 > 0.f ? x0 : (__expf(x0) - 1.f);             // ELU
    int b0 = bg * 32 + wb * 16 + m;
    hbf[(size_t)b0 * H_SZ + j] = f2bf(x0);
  }
}

// ---------------------------------------------------------------------------
// GEMM2 + dX contraction + fused RK4 stage update.
// out[b,hh] = sum_c dXw[b,c] * ( (h @ W2c^T)[b,hh] + bias2[hh*C+c]*Wv[c] )
// mode 0:   kacc  = w*k; zin = bf16(z + az*k)
// mode 1/2: kacc += w*k; zin = bf16(z + az*k)
// mode 3:   z += kacc + w*k; zin = bf16(z)
// grid (8,8): block = 32 b-rows x 64 hh-cols, 8 waves, 1 tile per wave.
// ---------------------------------------------------------------------------
__global__ __launch_bounds__(256) void gemm2_kernel(
    const unsigned short* __restrict__ hbf, const uint4* __restrict__ W2p,
    const float* __restrict__ cb, const float* __restrict__ cc, const float* __restrict__ cd,
    const float* __restrict__ bias2, const float* __restrict__ Wv,
    float* __restrict__ z, float* __restrict__ kacc, unsigned short* __restrict__ zin,
    int t, float s, int mode, float w_acc, float a_z) {
  __shared__ __align__(16) unsigned short ht[32 * 520];
  __shared__ float dxw[32][33];
  __shared__ float bbs[64][33];
  int bg = blockIdx.x, hg = blockIdx.y, tid = threadIdx.x;

  {
    const uint4* src = (const uint4*)hbf + (size_t)bg * 32 * 64;
    uint4* dst = (uint4*)ht;
    for (int i = tid; i < 2048; i += 256) {
      int r = i >> 6, cu = i & 63;
      dst[r * 65 + cu] = src[r * 64 + cu];
    }
  }
  for (int i = tid; i < 32 * 32; i += 256) {             // dX(s) per (b-row, c)
    int r = i >> 5, c = i & 31;
    int base = ((bg * 32 + r) * TM1 + t) * C_SZ + c;
    dxw[r][c] = cb[base] + s * (2.f * cc[base] + 3.f * s * cd[base]);
  }
  for (int i = tid; i < 64 * 32; i += 256) {             // bias2*Wv for this hh-range
    int hl = i >> 5, c = i & 31;
    bbs[hl][c] = bias2[(hg * 64 + hl) * C_SZ + c] * Wv[c];
  }
  __syncthreads();

  int lane = tid & 31, wave = tid >> 5;
  int wh = wave & 3, wb = wave >> 2;
  int half = lane >> 4, n = lane & 15;
  int hht = hg * 4 + wh;
  const uint4* lq = (const uint4*)ht;
  int arow = wb * 16 * 65;

  v8f zf = {0.f,0.f,0.f,0.f,0.f,0.f,0.f,0.f};
  Acc8 f0; f0.v = zf;

  for (int c = 0; c < C_SZ; ++c) {
    Acc8 accE, accO; accE.v = zf; accO.v = zf;           // dual chains
    const uint4* wbp = W2p + ((((size_t)(c * 32 + hht) * 16) * 32 + lane) << 1);
    #pragma unroll
    for (int kt = 0; kt < 16; kt += 2) {
      FragB bfr0, bfr1;
      bfr0.q[0] = wbp[(size_t)kt * 64];
      bfr0.q[1] = wbp[(size_t)kt * 64 + 1];
      bfr1.q[0] = wbp[(size_t)(kt + 1) * 64];
      bfr1.q[1] = wbp[(size_t)(kt + 1) * 64 + 1];
      FragB af0, af1;
      int ab = arow + n * 65 + kt * 4 + half * 2;
      af0.q[0] = lq[ab];     af0.q[1] = lq[ab + 1];
      af1.q[0] = lq[ab + 4]; af1.q[1] = lq[ab + 5];
      accE.v = WMMA_BF16(af0.v, bfr0.v, accE.v);
      accO.v = WMMA_BF16(af1.v, bfr1.v, accO.v);
    }
    float bbv = bbs[wh * 16 + n][c];
    #pragma unroll
    for (int v = 0; v < 8; ++v) {
      int m = wb * 16 + v + half * 8;
      f0.f[v] += dxw[m][c] * (accE.f[v] + accO.f[v] + bbv);
    }
  }

  int hh = hht * 16 + n;
  #pragma unroll
  for (int v = 0; v < 8; ++v) {
    int m = wb * 16 + v + half * 8;
    int b0 = bg * 32 + m;
    size_t i0 = (size_t)b0 * H_SZ + hh;
    float k0 = f0.f[v];
    if (mode == 0) {
      kacc[i0] = w_acc * k0;
      zin[i0] = f2bf(z[i0] + a_z * k0);
    } else if (mode == 3) {
      float zn0 = z[i0] + kacc[i0] + w_acc * k0;
      z[i0] = zn0;
      zin[i0] = f2bf(zn0);
    } else {
      kacc[i0] += w_acc * k0;
      zin[i0] = f2bf(z[i0] + a_z * k0);
    }
  }
}

// Readout: out[b,t,:] = z[b,:] @ Wr^T + br  (fp32, tiny)
__global__ void readout_kernel(const float* __restrict__ z, const float* __restrict__ Wr,
                               const float* __restrict__ br, float* __restrict__ out, int t) {
  int id = blockIdx.x * blockDim.x + threadIdx.x;        // B*O = 2048
  if (id >= B_SZ * O_SZ) return;
  int b = id >> 3, o = id & 7;
  float acc = br[o];
  const float* zr = z + (size_t)b * H_SZ;
  const float* wr = Wr + (size_t)o * H_SZ;
  for (int k = 0; k < H_SZ; ++k) acc += zr[k] * wr[k];
  out[((size_t)b * T_SZ + t) * O_SZ + o] = acc;
}

extern "C" void kernel_launch(void* const* d_in, const int* in_sizes, int n_in,
                              void* d_out, int out_size, void* d_ws, size_t ws_size,
                              hipStream_t stream) {
  (void)in_sizes; (void)n_in; (void)out_size; (void)ws_size;
  const float* ca    = (const float*)d_in[0];
  const float* cb    = (const float*)d_in[1];
  const float* cc    = (const float*)d_in[2];
  const float* cd    = (const float*)d_in[3];
  const float* W1    = (const float*)d_in[4];
  const float* bias1 = (const float*)d_in[5];
  const float* W2    = (const float*)d_in[6];
  const float* bias2 = (const float*)d_in[7];
  const float* Wv    = (const float*)d_in[8];
  const float* Wi    = (const float*)d_in[9];
  const float* bi    = (const float*)d_in[10];
  const float* Wr    = (const float*)d_in[11];
  const float* br    = (const float*)d_in[12];
  float* out = (float*)d_out;

  // Workspace layout (~18.9 MB)
  char* ws = (char*)d_ws;
  unsigned* W1p = (unsigned*)ws;                                       // 512 KB
  unsigned* W2p = (unsigned*)(ws + (512u << 10));                      // 16 MB
  float* z      = (float*)(ws + (512u << 10) + (16u << 20));           // 512 KB
  float* kacc   = (float*)((char*)z    + (512u << 10));                // 512 KB
  unsigned short* zin = (unsigned short*)((char*)kacc + (512u << 10)); // 256 KB
  unsigned short* hbf = (unsigned short*)((char*)zin  + (256u << 10)); // 256 KB

  pack_w1_kernel<<<131072 / 256, 256, 0, stream>>>(W1, W1p);
  pack_w2_kernel<<<4194304 / 256, 256, 0, stream>>>(W2, Wv, W2p);
  init_z0_kernel<<<131072 / 256, 256, 0, stream>>>(ca, Wi, bi, z, zin);
  readout_kernel<<<8, 256, 0, stream>>>(z, Wr, br, out, 0);

  const float ds = 0.5f;
  dim3 g(8, 8);
  for (int t = 0; t < TM1; ++t) {
    for (int j = 0; j < 2; ++j) {
      float s0 = j * ds;
      const float sv[4] = { s0, s0 + 0.5f * ds, s0 + 0.5f * ds, s0 + ds };
      const int   md[4] = { 0, 1, 2, 3 };
      const float wa[4] = { ds / 6.f, ds / 3.f, ds / 3.f, ds / 6.f };
      const float az[4] = { 0.5f * ds, 0.5f * ds, ds, 0.f };
      for (int e = 0; e < 4; ++e) {
        gemm1_kernel<<<g, 256, 0, stream>>>(zin, (const uint4*)W1p, bias1, hbf);
        gemm2_kernel<<<g, 256, 0, stream>>>(hbf, (const uint4*)W2p, cb, cc, cd,
                                            bias2, Wv, z, kacc, zin,
                                            t, sv[e], md[e], wa[e], az[e]);
      }
    }
    readout_kernel<<<8, 256, 0, stream>>>(z, Wr, br, out, t + 1);
  }
}
